// GCCGraphInfer_48215302864931
// MI455X (gfx1250) — compile-verified, matrix operand order checked by reading, not run on hardware
//
#include <hip/hip_runtime.h>
#include <math.h>

typedef __bf16 bf16;
typedef __attribute__((ext_vector_type(16))) bf16  v16bf;
typedef __attribute__((ext_vector_type(8)))  bf16  v8bf;
typedef __attribute__((ext_vector_type(8)))  float v8f;

#define NN 50000
#define EE 400000
#define GG 64

// ---------------- helpers ----------------

__device__ __forceinline__ unsigned fenc(float f) {
  unsigned u = __float_as_uint(f);
  return (u & 0x80000000u) ? ~u : (u | 0x80000000u);
}
__device__ __forceinline__ float fdec(unsigned u) {
  return (u & 0x80000000u) ? __uint_as_float(u & 0x7FFFFFFFu) : __uint_as_float(~u);
}
__device__ __forceinline__ float lrelu02(float x) { return x > 0.f ? x : 0.2f * x; }

// Load one 16x32 bf16 WMMA fragment (A row-major, or B from transposed Bt[N][K]).
// p = row base + k0 ; half = lane>>4.
__device__ __forceinline__ v16bf load_frag_bf16(const bf16* p, int half) {
  v8bf lo = *(const v8bf*)(p + half * 8);
  v8bf hi = *(const v8bf*)(p + 16 + half * 8);
  v16bf f;
#pragma unroll
  for (int i = 0; i < 8; ++i) { f[i] = lo[i]; f[i + 8] = hi[i]; }
  return f;
}

// ---------------- conversion / transpose ----------------

__global__ void conv_bf16_kernel(const float* __restrict__ in, bf16* __restrict__ out, long n) {
  long i = (long)blockIdx.x * blockDim.x + threadIdx.x;
  if (i < n) out[i] = (bf16)in[i];
}

// W[K,Nt] f32 -> Wt[Nt,K] bf16
__global__ void transpose_w_kernel(const float* __restrict__ W, bf16* __restrict__ Wt,
                                   int K, int Nt) {
  int i = blockIdx.x * blockDim.x + threadIdx.x;
  if (i >= K * Nt) return;
  int k = i / Nt, n = i % Nt;
  Wt[(size_t)n * K + k] = (bf16)W[i];
}

// ---------------- WMMA GEMM:  C = A[M,K] * B[K,Nt] (+bias) ----------------
// A: bf16 row-major. Bt: bf16 [Nt,K] (transposed). MODE 0: store f32 out[M,Nt].
// MODE 1: tanh(acc+bias) then atomicAdd column sums into out[Nt].
template <int MODE>
__global__ __launch_bounds__(256) void gemm_bf16_wmma(
    const bf16* __restrict__ A, const bf16* __restrict__ Bt,
    const float* __restrict__ bias, float* __restrict__ out,
    int M, int K, int Nt) {
  int wave = blockIdx.x * (blockDim.x >> 5) + (threadIdx.x >> 5);
  int lane = threadIdx.x & 31;
  int tilesM = M >> 4, tilesN = Nt >> 4;
  if (wave >= tilesM * tilesN) return;           // wave-uniform: EXEC stays full
  int tm = wave % tilesM, tn = wave / tilesM;
  int half = lane >> 4, r = lane & 15;

  const bf16* arow = A  + (size_t)(tm * 16 + r) * K;
  const bf16* brow = Bt + (size_t)(tn * 16 + r) * K;

  v8f acc = {0.f, 0.f, 0.f, 0.f, 0.f, 0.f, 0.f, 0.f};
  for (int k0 = 0; k0 < K; k0 += 32) {
    v16bf af = load_frag_bf16(arow + k0, half);
    v16bf bf = load_frag_bf16(brow + k0, half);
    acc = __builtin_amdgcn_wmma_f32_16x16x32_bf16(false, af, false, bf,
                                                  (short)0, acc, false, false);
  }

  int col = tn * 16 + r;
  float b = bias[col];
  if (MODE == 0) {
#pragma unroll
    for (int i = 0; i < 8; ++i) {
      int row = tm * 16 + half * 8 + i;
      out[(size_t)row * Nt + col] = acc[i] + b;
    }
  } else {
    float s = 0.f;
#pragma unroll
    for (int i = 0; i < 8; ++i) s += tanhf(acc[i] + b);
    atomicAdd(&out[col], s);
  }
}

// ---------------- attention (edge) kernels ----------------

// s[n,h] = sum_d h[n, h*D+d]*att_src[h,d]; d likewise
__global__ void sd_kernel(const float* __restrict__ h, const float* __restrict__ att_src,
                          const float* __restrict__ att_dst, float* __restrict__ sb,
                          float* __restrict__ db, int Nn, int D) {
  int idx = blockIdx.x * blockDim.x + threadIdx.x;
  if (idx >= Nn * 4) return;
  int n = idx >> 2, hh = idx & 3;
  const float* hp = h + (size_t)n * 4 * D + hh * D;
  const float* as = att_src + hh * D;
  const float* ad = att_dst + hh * D;
  float ss = 0.f, dd = 0.f;
  for (int j = 0; j < D; ++j) { float v = hp[j]; ss += v * as[j]; dd += v * ad[j]; }
  sb[idx] = ss; db[idx] = dd;
}

__global__ void edge_max_kernel(const int* __restrict__ src, const int* __restrict__ dst,
                                const float* __restrict__ sb, const float* __restrict__ db,
                                unsigned* __restrict__ amax, int Ecount) {
  int idx = blockIdx.x * blockDim.x + threadIdx.x;
  if (idx >= Ecount * 4) return;
  int e = idx >> 2, hh = idx & 3;
  int si = src[e], di = dst[e];
  float a = lrelu02(sb[si * 4 + hh] + db[di * 4 + hh]);
  atomicMax(&amax[di * 4 + hh], fenc(a));
}

__global__ void edge_exp_kernel(const int* __restrict__ src, const int* __restrict__ dst,
                                const float* __restrict__ sb, const float* __restrict__ db,
                                const unsigned* __restrict__ amax, float* __restrict__ exbuf,
                                float* __restrict__ denom, int Ecount) {
  int idx = blockIdx.x * blockDim.x + threadIdx.x;
  if (idx >= Ecount * 4) return;
  int e = idx >> 2, hh = idx & 3;
  int si = src[e], di = dst[e];
  float a = lrelu02(sb[si * 4 + hh] + db[di * 4 + hh]);
  float ex = expf(a - fdec(amax[di * 4 + hh]));
  exbuf[idx] = ex;
  atomicAdd(&denom[di * 4 + hh], ex);
}

// out[dst, f] += h[src, f] * ex/denom   (F = 1<<logF, D = 1<<logD)
__global__ void edge_aggr_kernel(const int* __restrict__ src, const int* __restrict__ dst,
                                 const float* __restrict__ h, const float* __restrict__ exbuf,
                                 const float* __restrict__ denom, float* __restrict__ out,
                                 int Ecount, int logF, int logD) {
  int idx = blockIdx.x * blockDim.x + threadIdx.x;
  int F = 1 << logF;
  if (idx >= (Ecount << logF)) return;
  int e = idx >> logF, f = idx & (F - 1);
  int hh = f >> logD;
  int si = src[e], di = dst[e];
  float w = exbuf[e * 4 + hh] / (denom[di * 4 + hh] + 1e-16f);
  atomicAdd(&out[(size_t)di * F + f], h[(size_t)si * F + f] * w);
}

// relu in place + bf16 copy (klin GEMM input)
__global__ void relu_conv_kernel(float* __restrict__ buf, bf16* __restrict__ obf, long n) {
  long i = (long)blockIdx.x * blockDim.x + threadIdx.x;
  if (i >= n) return;
  float v = buf[i];
  v = v > 0.f ? v : 0.f;
  buf[i] = v;
  obf[i] = (bf16)v;
}

// semantic attention: w_r = colsum_r / N ; z_r = w_r . q ; attn = softmax over R=2
__global__ void semattn_kernel(const float* __restrict__ colsum, const float* __restrict__ q,
                               float* __restrict__ attn, int F, float invN) {
  __shared__ float red[256];
  int t = threadIdx.x;
  float p0 = 0.f, p1 = 0.f;
  for (int f = t; f < F; f += 256) {
    p0 += colsum[f] * invN * q[f];
    p1 += colsum[F + f] * invN * q[f];
  }
  red[t] = p0; __syncthreads();
  for (int s = 128; s > 0; s >>= 1) { if (t < s) red[t] += red[t + s]; __syncthreads(); }
  float z0 = red[0]; __syncthreads();
  red[t] = p1; __syncthreads();
  for (int s = 128; s > 0; s >>= 1) { if (t < s) red[t] += red[t + s]; __syncthreads(); }
  float z1 = red[0];
  if (t == 0) {
    float m = fmaxf(z0, z1);
    float e0 = expf(z0 - m), e1 = expf(z1 - m);
    float inv = 1.f / (e0 + e1);
    attn[0] = e0 * inv; attn[1] = e1 * inv;
  }
}

__global__ void combine_bf16_kernel(const float* __restrict__ r0, const float* __restrict__ r1,
                                    const float* __restrict__ attn, bf16* __restrict__ out, long n) {
  long i = (long)blockIdx.x * blockDim.x + threadIdx.x;
  if (i >= n) return;
  out[i] = (bf16)(attn[0] * r0[i] + attn[1] * r1[i]);
}
__global__ void combine_f32_kernel(const float* __restrict__ r0, const float* __restrict__ r1,
                                   const float* __restrict__ attn, float* __restrict__ out, long n) {
  long i = (long)blockIdx.x * blockDim.x + threadIdx.x;
  if (i >= n) return;
  out[i] = attn[0] * r0[i] + attn[1] * r1[i];
}

// ---------------- pooling + MLP head ----------------

__global__ void pool_kernel(const float* __restrict__ h, const int* __restrict__ batch,
                            float* __restrict__ pooled, float* __restrict__ cnt,
                            int Nn, int F) {
  int idx = blockIdx.x * blockDim.x + threadIdx.x;
  if (idx >= Nn * F) return;
  int n = idx / F, f = idx - n * F;
  int g = batch[n];
  atomicAdd(&pooled[(size_t)g * F + f], h[idx]);
  if (f == 0) atomicAdd(&cnt[g], 1.0f);
}

__global__ void head_kernel(const float* __restrict__ pooled, const float* __restrict__ cnt,
                            const float* __restrict__ w1, const float* __restrict__ b1,
                            const float* __restrict__ gamma, const float* __restrict__ beta,
                            const float* __restrict__ mean, const float* __restrict__ var,
                            const float* __restrict__ w2, const float* __restrict__ b2,
                            float* __restrict__ out, int Gn) {
  int g = threadIdx.x;
  if (g >= Gn) return;
  float c = cnt[g]; c = c > 1.0f ? c : 1.0f;
  float inv = 1.0f / c;
  float acc = b2[0];
  for (int j = 0; j < 64; ++j) {
    float z = b1[j];
    for (int k = 0; k < 128; ++k) z += (pooled[g * 128 + k] * inv) * w1[k * 64 + j];
    z = (z - mean[j]) * rsqrtf(var[j] + 1e-5f) * gamma[j] + beta[j];
    z = z > 0.f ? z : 0.1f * z;
    acc += z * w2[j];
  }
  out[g] = acc;
}

// ---------------- host-side layer driver ----------------

static void run_han_layer(hipStream_t stream, int Nn, int Ecount, int F, int D,
                          int logF, int logD,
                          const float* h, const int* ei0, const int* ei1,
                          const float* as0, const float* ad0,
                          const float* as1, const float* ad1,
                          const bf16* klin_wt, const float* klin_b, const float* q,
                          float* sbuf, float* dbuf, unsigned* amax, float* denom,
                          float* exbuf, float* relu0, float* relu1,
                          bf16* scratch_bf, float* colsum, float* attnbuf,
                          bf16* out_bf, float* out_f32) {
  const int* eis[2] = {ei0, ei1};
  const float* asv[2] = {as0, as1};
  const float* adv[2] = {ad0, ad1};
  float* relu[2] = {relu0, relu1};

  hipMemsetAsync(relu0, 0, (size_t)Nn * F * 4, stream);
  hipMemsetAsync(relu1, 0, (size_t)Nn * F * 4, stream);
  hipMemsetAsync(colsum, 0, (size_t)2 * F * 4, stream);

  long nf = (long)Nn * F;
  int tiles = (Nn >> 4) * (F >> 4);

  for (int r = 0; r < 2; ++r) {
    const int* src = eis[r];
    const int* dst = eis[r] + Ecount;
    sd_kernel<<<(Nn * 4 + 255) / 256, 256, 0, stream>>>(h, asv[r], adv[r], sbuf, dbuf, Nn, D);
    hipMemsetAsync(amax, 0, (size_t)Nn * 4 * 4, stream);
    hipMemsetAsync(denom, 0, (size_t)Nn * 4 * 4, stream);
    edge_max_kernel<<<(Ecount * 4 + 255) / 256, 256, 0, stream>>>(src, dst, sbuf, dbuf, amax, Ecount);
    edge_exp_kernel<<<(Ecount * 4 + 255) / 256, 256, 0, stream>>>(src, dst, sbuf, dbuf, amax,
                                                                  exbuf, denom, Ecount);
    long ef = (long)Ecount << logF;
    edge_aggr_kernel<<<(int)((ef + 255) / 256), 256, 0, stream>>>(src, dst, h, exbuf, denom,
                                                                  relu[r], Ecount, logF, logD);
    relu_conv_kernel<<<(int)((nf + 255) / 256), 256, 0, stream>>>(relu[r], scratch_bf, nf);
    gemm_bf16_wmma<1><<<(tiles + 7) / 8, 256, 0, stream>>>(scratch_bf, klin_wt, klin_b,
                                                           colsum + r * F, Nn, F, F);
  }
  semattn_kernel<<<1, 256, 0, stream>>>(colsum, q, attnbuf, F, 1.0f / (float)Nn);
  if (out_bf)
    combine_bf16_kernel<<<(int)((nf + 255) / 256), 256, 0, stream>>>(relu0, relu1, attnbuf, out_bf, nf);
  else
    combine_f32_kernel<<<(int)((nf + 255) / 256), 256, 0, stream>>>(relu0, relu1, attnbuf, out_f32, nf);
}

// ---------------- entry point ----------------

extern "C" void kernel_launch(void* const* d_in, const int* in_sizes, int n_in,
                              void* d_out, int out_size, void* d_ws, size_t ws_size,
                              hipStream_t stream) {
  (void)in_sizes; (void)n_in; (void)out_size; (void)ws_size;

  const float* x       = (const float*)d_in[0];
  const int*   ei0     = (const int*)d_in[1];
  const int*   ei1     = (const int*)d_in[2];
  const int*   batch   = (const int*)d_in[3];
  const float* proj1_w = (const float*)d_in[4];
  const float* proj1_b = (const float*)d_in[5];
  const float* as10    = (const float*)d_in[6];
  const float* ad10    = (const float*)d_in[7];
  const float* as11    = (const float*)d_in[8];
  const float* ad11    = (const float*)d_in[9];
  const float* klin1_w = (const float*)d_in[10];
  const float* klin1_b = (const float*)d_in[11];
  const float* q1      = (const float*)d_in[12];
  const float* proj2_w = (const float*)d_in[13];
  const float* proj2_b = (const float*)d_in[14];
  const float* as20    = (const float*)d_in[15];
  const float* ad20    = (const float*)d_in[16];
  const float* as21    = (const float*)d_in[17];
  const float* ad21    = (const float*)d_in[18];
  const float* klin2_w = (const float*)d_in[19];
  const float* klin2_b = (const float*)d_in[20];
  const float* q2      = (const float*)d_in[21];
  const float* d1_w    = (const float*)d_in[22];
  const float* d1_b    = (const float*)d_in[23];
  const float* bn_g    = (const float*)d_in[24];
  const float* bn_b    = (const float*)d_in[25];
  const float* bn_m    = (const float*)d_in[26];
  const float* bn_v    = (const float*)d_in[27];
  const float* d2_w    = (const float*)d_in[28];
  const float* d2_b    = (const float*)d_in[29];

  // ---- workspace arena ----
  char* ws = (char*)d_ws;
  size_t off = 0;
  auto alloc = [&](size_t bytes) -> void* {
    void* p = ws + off;
    off += (bytes + 255) & ~(size_t)255;
    return p;
  };
  bf16*     x_bf       = (bf16*)alloc((size_t)NN * 128 * 2);
  bf16*     w_proj1_t  = (bf16*)alloc((size_t)256 * 128 * 2);
  bf16*     w_klin1_t  = (bf16*)alloc((size_t)256 * 256 * 2);
  bf16*     w_proj2_t  = (bf16*)alloc((size_t)128 * 256 * 2);
  bf16*     w_klin2_t  = (bf16*)alloc((size_t)128 * 128 * 2);
  float*    h1         = (float*)alloc((size_t)NN * 256 * 4);   // also reused as h2
  float*    relu0      = (float*)alloc((size_t)NN * 256 * 4);
  float*    relu1      = (float*)alloc((size_t)NN * 256 * 4);
  bf16*     scratch_bf = (bf16*)alloc((size_t)NN * 256 * 2);    // also reused as hcomb2 (f32 N*128)
  bf16*     hcomb1_bf  = (bf16*)alloc((size_t)NN * 256 * 2);
  float*    sbuf       = (float*)alloc((size_t)NN * 4 * 4);
  float*    dbuf       = (float*)alloc((size_t)NN * 4 * 4);
  unsigned* amax       = (unsigned*)alloc((size_t)NN * 4 * 4);
  float*    denom      = (float*)alloc((size_t)NN * 4 * 4);
  float*    exbuf      = (float*)alloc((size_t)EE * 4 * 4);
  float*    colsum     = (float*)alloc((size_t)2 * 256 * 4);
  float*    attnbuf    = (float*)alloc((size_t)2 * 4);
  float*    pooled     = (float*)alloc((size_t)GG * 128 * 4);
  float*    cnt        = (float*)alloc((size_t)GG * 4);

  // ---- weight prep (bf16 + transpose) ----
  conv_bf16_kernel<<<(NN * 128 + 255) / 256, 256, 0, stream>>>(x, x_bf, (long)NN * 128);
  transpose_w_kernel<<<(128 * 256 + 255) / 256, 256, 0, stream>>>(proj1_w, w_proj1_t, 128, 256);
  transpose_w_kernel<<<(256 * 256 + 255) / 256, 256, 0, stream>>>(klin1_w, w_klin1_t, 256, 256);
  transpose_w_kernel<<<(256 * 128 + 255) / 256, 256, 0, stream>>>(proj2_w, w_proj2_t, 256, 128);
  transpose_w_kernel<<<(128 * 128 + 255) / 256, 256, 0, stream>>>(klin2_w, w_klin2_t, 128, 128);

  // ---- layer 1: proj (128 -> 256) via WMMA, then HAN edge phase ----
  {
    int tiles = (NN >> 4) * (256 >> 4);
    gemm_bf16_wmma<0><<<(tiles + 7) / 8, 256, 0, stream>>>(x_bf, w_proj1_t, proj1_b, h1,
                                                           NN, 128, 256);
  }
  run_han_layer(stream, NN, EE, 256, 64, 8, 6, h1, ei0, ei1, as10, ad10, as11, ad11,
                w_klin1_t, klin1_b, q1, sbuf, dbuf, amax, denom, exbuf,
                relu0, relu1, scratch_bf, colsum, attnbuf, hcomb1_bf, nullptr);

  // ---- layer 2: proj (256 -> 128) via WMMA, then HAN edge phase ----
  {
    int tiles = (NN >> 4) * (128 >> 4);
    gemm_bf16_wmma<0><<<(tiles + 7) / 8, 256, 0, stream>>>(hcomb1_bf, w_proj2_t, proj2_b, h1,
                                                           NN, 256, 128);
  }
  float* hcomb2 = (float*)scratch_bf;  // aliases scratch, safe after klin gemms
  run_han_layer(stream, NN, EE, 128, 32, 7, 5, h1, ei0, ei1, as20, ad20, as21, ad21,
                w_klin2_t, klin2_b, q2, sbuf, dbuf, amax, denom, exbuf,
                relu0, relu1, scratch_bf, colsum, attnbuf, nullptr, hcomb2);

  // ---- global mean pool + MLP head ----
  hipMemsetAsync(pooled, 0, (size_t)GG * 128 * 4, stream);
  hipMemsetAsync(cnt, 0, (size_t)GG * 4, stream);
  pool_kernel<<<(NN * 128 + 255) / 256, 256, 0, stream>>>(hcomb2, batch, pooled, cnt, NN, 128);
  head_kernel<<<1, 64, 0, stream>>>(pooled, cnt, d1_w, d1_b, bn_g, bn_b, bn_m, bn_v,
                                    d2_w, d2_b, (float*)d_out, GG);
}
